// ArcNegFace_31739808317518
// MI455X (gfx1250) — matrix operand order repaired and checked by reading, not compiled, not run on hardware
//
#include <hip/hip_runtime.h>

#define B_ROWS  256
#define D_DIM   512
#define C_CLS   100000
#define N_TILES (C_CLS / 16)          // 6250
#define RS      520                    // padded LDS row stride in bf16 elems (breaks bank conflicts)
#define SCALE_F 64.0f
#define ALPHA_F 1.2f
#define MARGIN_F 0.5f
#define THRESH_F (-0.8775825618903728f)   // cos(pi - 0.5)
#define MM_F     (0.2397127693021015f)    // sin(pi - 0.5) * 0.5

// ---- optional gfx1250 async global->LDS path (guarded; falls back cleanly) ----
#ifndef HAVE_ASYNC_LDS
#if defined(__has_builtin)
#if __has_builtin(__builtin_amdgcn_global_load_async_to_lds_b128)
#define HAVE_ASYNC_LDS 1
#endif
#endif
#endif
#ifndef HAVE_ASYNC_LDS
#define HAVE_ASYNC_LDS 0
#endif

#define AS1 __attribute__((address_space(1)))
#define AS3 __attribute__((address_space(3)))

typedef __attribute__((ext_vector_type(16))) __bf16 v16bf;
typedef __attribute__((ext_vector_type(8)))  __bf16 v8bf;
typedef __attribute__((ext_vector_type(8)))  float  v8f;
typedef __attribute__((ext_vector_type(4)))  int    v4i;

union V16U { v16bf v; v8bf h[2]; };

static __device__ __forceinline__ unsigned int f2bf(float f) {
    unsigned int u = __float_as_uint(f);
    u += 0x7FFFu + ((u >> 16) & 1u);      // round-to-nearest-even
    return u >> 16;
}

static __device__ __forceinline__ float wave_sum(float v) {
    #pragma unroll
    for (int o = 16; o > 0; o >>= 1) v += __shfl_xor(v, o, 32);
    return v;
}

#if HAVE_ASYNC_LDS
static __device__ __forceinline__ void wait_async0() {
#if defined(__has_builtin) && __has_builtin(__builtin_amdgcn_s_wait_asynccnt)
    __builtin_amdgcn_s_wait_asynccnt(0);
#else
    asm volatile("s_wait_asynccnt 0x0" ::: "memory");
#endif
}
#endif

// ---------------- Kernel 1: normalize input rows -> bf16 ----------------
__global__ __launch_bounds__(256) void innorm_kernel(const float* __restrict__ in,
                                                     unsigned short* __restrict__ outbf) {
    int wid = threadIdx.x >> 5, lane = threadIdx.x & 31;
    int row = blockIdx.x * 8 + wid;                       // 32 blocks * 8 waves = 256 rows
    const float4* p = (const float4*)(in + (size_t)row * D_DIM);
    float4 v[4];
    float ss = 0.f;
    #pragma unroll
    for (int j = 0; j < 4; ++j) {
        v[j] = p[lane + 32 * j];
        ss += v[j].x * v[j].x + v[j].y * v[j].y + v[j].z * v[j].z + v[j].w * v[j].w;
    }
    ss = wave_sum(ss);
    float inv = rsqrtf(fmaxf(ss, 1e-24f));
    uint2* dst = (uint2*)(outbf + (size_t)row * D_DIM);
    #pragma unroll
    for (int j = 0; j < 4; ++j) {
        uint2 u;
        u.x = f2bf(v[j].x * inv) | (f2bf(v[j].y * inv) << 16);
        u.y = f2bf(v[j].z * inv) | (f2bf(v[j].w * inv) << 16);
        dst[lane + 32 * j] = u;
    }
}

// ---------------- Kernel 2: per-row margined target logit a_lb ----------------
__global__ __launch_bounds__(256) void alb_kernel(const float* __restrict__ in,
                                                  const int* __restrict__ label,
                                                  const float* __restrict__ w,
                                                  float* __restrict__ alb) {
    int wid = threadIdx.x >> 5, lane = threadIdx.x & 31;
    int m = blockIdx.x * 8 + wid;
    int lbl = label[m];
    const float4* pi = (const float4*)(in + (size_t)m * D_DIM);
    const float4* pw = (const float4*)(w + (size_t)lbl * D_DIM);
    float ss = 0.f, dp = 0.f, sw = 0.f;
    #pragma unroll
    for (int j = 0; j < 4; ++j) {
        float4 a = pi[lane + 32 * j];
        float4 b = pw[lane + 32 * j];
        ss += a.x * a.x + a.y * a.y + a.z * a.z + a.w * a.w;
        sw += b.x * b.x + b.y * b.y + b.z * b.z + b.w * b.w;
        dp += a.x * b.x + a.y * b.y + a.z * b.z + a.w * b.w;
    }
    ss = wave_sum(ss);
    sw = wave_sum(sw);
    dp = wave_sum(dp);
    if (lane == 0) {
        float cl = dp * rsqrtf(fmaxf(ss, 1e-24f)) * rsqrtf(fmaxf(sw, 1e-24f));
        float cc = fminf(1.f, fmaxf(-1.f, cl));
        float a  = (cl > THRESH_F) ? cosf(acosf(cc) + MARGIN_F) : (cl - MM_F);
        alb[m] = a;
    }
}

// ---------------- Kernel 3: bf16 WMMA GEMM + fused weight-norm + epilogue ----
// 8 waves; bf16 input (256x520 padded, ~260KB) resident in LDS via async copy;
// weight tiles of 16 classes double-buffered (register prefetch -> wave32
// sum-of-squares reduction -> bf16 -> LDS). Weight is read from HBM ONCE.
__global__ __launch_bounds__(256, 1) void gemm_kernel(const float* __restrict__ weight,
                                                      const unsigned short* __restrict__ in_n,
                                                      const float* __restrict__ alb,
                                                      const int* __restrict__ label,
                                                      float* __restrict__ out) {
    extern __shared__ unsigned char smem_raw[];
    unsigned short* in_s = (unsigned short*)smem_raw;            // 256*520 bf16
    unsigned short* w_s  = in_s + B_ROWS * RS;                   // 2*16*520 bf16
    float* alb_s  = (float*)(w_s + 2 * 16 * RS);                 // 256 f32
    int*   lbl_s  = (int*)(alb_s + B_ROWS);                      // 256 i32
    float* winv_s = (float*)(lbl_s + B_ROWS);                    // 2*16 f32 (1/||w|| per buffer)

    const int tid  = threadIdx.x;
    const int lane = tid & 31;
    const int wid  = tid >> 5;
    const int hf   = lane >> 4;       // 0: lanes 0-15, 1: lanes 16-31
    const int nn   = lane & 15;       // class-in-tile / row-in-mtile selector
    const int r0   = wid * 2;         // this wave owns weight rows r0, r0+1 of each tile

    // ---- Stage normalized bf16 input into LDS (async global->LDS, 16B chunks) ----
    for (int j = 0; j < 64; ++j) {
        int c = tid + 256 * j;
        int row = c >> 6, col8 = c & 63;
#if HAVE_ASYNC_LDS
        __builtin_amdgcn_global_load_async_to_lds_b128(
            (AS1 v4i*)(in_n + (size_t)row * D_DIM + col8 * 8),
            (AS3 v4i*)(in_s + row * RS + col8 * 8), 0, 0);
#else
        uint4 v = ((const uint4*)(in_n + (size_t)row * D_DIM))[col8];
        *(uint4*)(in_s + row * RS + col8 * 8) = v;
#endif
    }
    if (tid < B_ROWS) { alb_s[tid] = alb[tid]; lbl_s[tid] = label[tid]; }
#if HAVE_ASYNC_LDS
    wait_async0();
#endif

    const int t0 = blockIdx.x, gs = gridDim.x;
    float4 wr[2][4];                  // register-prefetched weight tile slice (2 rows)

    // Load this wave's 2 rows of weight tile t into registers (coalesced 512B runs).
    auto load_tile = [&](int t) {
        #pragma unroll
        for (int rr = 0; rr < 2; ++rr) {
            const float4* src = (const float4*)(weight + ((size_t)t * 16 + r0 + rr) * D_DIM);
            #pragma unroll
            for (int q = 0; q < 4; ++q) wr[rr][q] = src[lane + 32 * q];
        }
    };
    // Fused: wave-reduce row sum-of-squares -> 1/||w|| to LDS, convert row to bf16 -> LDS.
    auto commit_tile = [&](int buf) {
        unsigned short* wn = w_s + buf * 16 * RS;
        #pragma unroll
        for (int rr = 0; rr < 2; ++rr) {
            float ssw = 0.f;
            #pragma unroll
            for (int q = 0; q < 4; ++q) {
                float4 v = wr[rr][q];
                ssw += v.x * v.x + v.y * v.y + v.z * v.z + v.w * v.w;
            }
            ssw = wave_sum(ssw);
            #pragma unroll
            for (int q = 0; q < 4; ++q) {
                uint2 u;
                u.x = f2bf(wr[rr][q].x) | (f2bf(wr[rr][q].y) << 16);
                u.y = f2bf(wr[rr][q].z) | (f2bf(wr[rr][q].w) << 16);
                *(uint2*)(wn + (r0 + rr) * RS + (lane + 32 * q) * 4) = u;
            }
            if (lane == 0) winv_s[buf * 16 + r0 + rr] = rsqrtf(fmaxf(ssw, 1e-24f));
        }
    };

    // Prologue: tile t0 into buffer 0.
    if (t0 < N_TILES) { load_tile(t0); commit_tile(0); }
    __syncthreads();

    int cur = 0;
    for (int t = t0; t < N_TILES; t += gs) {
        const int tn = t + gs;
        const bool hasnext = tn < N_TILES;

        if (hasnext) load_tile(tn);                       // register prefetch (1 tile ahead)
        if (t + 2 * gs < N_TILES)                         // L2 prefetch (2 tiles ahead)
            __builtin_prefetch(weight + ((size_t)(t + 2 * gs) * 16 + r0 + hf) * D_DIM + nn * 32,
                               0, 1);

        // ---- WMMA compute on current buffer: 2 M-tiles per wave, K=512 ----
        const unsigned short* wb = w_s + cur * 16 * RS;
        const unsigned short* pw = wb + nn * RS + hf * 8;                      // B frag base
        const unsigned short* pa = in_s + (wid * 16 + nn) * RS + hf * 8;       // A frag, mtile wid
        const unsigned short* pb = in_s + ((wid + 8) * 16 + nn) * RS + hf * 8; // A frag, mtile wid+8

        v8f acc0 = {}; v8f acc1 = {};
        #pragma unroll
        for (int kk = 0; kk < 16; ++kk) {
            const int ko = kk * 32;
            V16U bv, av, cv;
            bv.h[0] = *(const v8bf*)(pw + ko);      bv.h[1] = *(const v8bf*)(pw + ko + 16);
            av.h[0] = *(const v8bf*)(pa + ko);      av.h[1] = *(const v8bf*)(pa + ko + 16);
            cv.h[0] = *(const v8bf*)(pb + ko);      cv.h[1] = *(const v8bf*)(pb + ko + 16);
            acc0 = __builtin_amdgcn_wmma_f32_16x16x32_bf16(false, av.v, false, bv.v,
                                                           (short)0, acc0, false, false);
            acc1 = __builtin_amdgcn_wmma_f32_16x16x32_bf16(false, cv.v, false, bv.v,
                                                           (short)0, acc1, false, false);
        }

        // ---- Arc-margin epilogue + store (C/D layout: cls = lane&15, M = r + 8*hf) ----
        const int cls = t * 16 + nn;
        const float iw = winv_s[cur * 16 + nn];
        #pragma unroll
        for (int r = 0; r < 8; ++r) {
            {
                int m = wid * 16 + r + 8 * hf;
                float cv0 = acc0[r] * iw;
                float a = alb_s[m];
                float o;
                if (cls == lbl_s[m]) o = SCALE_F * a;
                else {
                    float dd = cv0 - a;
                    float ts = ALPHA_F * __expf(-dd * dd * 0.5f);
                    o = SCALE_F * (ts * cv0 + ts - 1.f);
                }
                out[(size_t)m * C_CLS + cls] = o;
            }
            {
                int m = (wid + 8) * 16 + r + 8 * hf;
                float cv1 = acc1[r] * iw;
                float a = alb_s[m];
                float o;
                if (cls == lbl_s[m]) o = SCALE_F * a;
                else {
                    float dd = cv1 - a;
                    float ts = ALPHA_F * __expf(-dd * dd * 0.5f);
                    o = SCALE_F * (ts * cv1 + ts - 1.f);
                }
                out[(size_t)m * C_CLS + cls] = o;
            }
        }

        // Commit prefetched tile (bf16 + fused norms) into the other buffer.
        if (hasnext) commit_tile(cur ^ 1);
        __syncthreads();
        cur ^= 1;
    }
}

// ---------------- Launch ----------------
// ws layout: [0, 1024) a_lb (256 f32); [1024, 263168) normalized input bf16 (256*512)
extern "C" void kernel_launch(void* const* d_in, const int* in_sizes, int n_in,
                              void* d_out, int out_size, void* d_ws, size_t ws_size,
                              hipStream_t stream) {
    const float* input  = (const float*)d_in[0];
    const int*   label  = (const int*)d_in[1];
    const float* weight = (const float*)d_in[2];
    float* out = (float*)d_out;

    float* alb           = (float*)d_ws;
    unsigned short* in_n = (unsigned short*)((char*)d_ws + 1024);

    hipLaunchKernelGGL(innorm_kernel, dim3(B_ROWS / 8), dim3(256), 0, stream, input, in_n);
    hipLaunchKernelGGL(alb_kernel, dim3(B_ROWS / 8), dim3(256), 0, stream,
                       input, label, weight, alb);

    const size_t smem = (size_t)(B_ROWS * RS + 2 * 16 * RS) * 2  // bf16 input + weight dbl-buf
                      + (size_t)B_ROWS * 4                        // a_lb
                      + (size_t)B_ROWS * 4                        // labels
                      + 2 * 16 * 4;                               // fused 1/||w||
    hipLaunchKernelGGL(gemm_kernel, dim3(512), dim3(256), smem, stream,
                       weight, in_n, alb, label, out);
}
